// QuantumTransformerE2E_83863531421722
// MI455X (gfx1250) — matrix operand order repaired and checked by reading, not compiled, not run on hardware
//
#include <hip/hip_runtime.h>
#include <hip/hip_bf16.h>
#include <math.h>

typedef __attribute__((ext_vector_type(16))) _Float16 v16h;
typedef __attribute__((ext_vector_type(8)))  float    v8f;

#define NQ      6
#define DM      128
#define CH      4
#define SEQ     2048
#define CHUNK   16
#define BATCH   128
#define NCHUNK  (SEQ / CHUNK)          // 128
#define NROWS   (BATCH * SEQ)          // 262144
#define NCH_TOT (BATCH * NCHUNK)       // 16384
#define NFP     60
#define QOUT    (3 * NQ)               // 18
#define PI_F    3.14159265358979f

union AFrag { v16h v; _Float16 e[16]; };
union Acc   { v8f  v; float e[8]; };

// ---------------------------------------------------------------------------
// Kernel 1: Conv1d(4->128, k=3, pad=1) + bias + LayerNorm -> h (f16)
// ---------------------------------------------------------------------------
__global__ void k_conv_ln(const float* __restrict__ x, const float* __restrict__ cw,
                          const float* __restrict__ cb, const float* __restrict__ g,
                          const float* __restrict__ beta, _Float16* __restrict__ hf16) {
  int pos = blockIdx.x;             // 0..NROWS-1
  int b = pos / SEQ, s = pos % SEQ;
  int o = threadIdx.x;              // 0..127
  float v = cb[o];
  const float* xb = x + (size_t)b * CH * SEQ;
#pragma unroll
  for (int c = 0; c < CH; ++c) {
    const float* xc = xb + c * SEQ;
    const float* w  = cw + (o * CH + c) * 3;
    float x0 = (s > 0)       ? xc[s - 1] : 0.f;
    float x1 =                 xc[s];
    float x2 = (s < SEQ - 1) ? xc[s + 1] : 0.f;
    v += x0 * w[0] + x1 * w[1] + x2 * w[2];
  }
  __shared__ float red[DM];
  red[o] = v; __syncthreads();
  for (int off = 64; off > 0; off >>= 1) { if (o < off) red[o] += red[o + off]; __syncthreads(); }
  float mean = red[0] * (1.f / DM);
  __syncthreads();
  float d = v - mean;
  red[o] = d * d; __syncthreads();
  for (int off = 64; off > 0; off >>= 1) { if (o < off) red[o] += red[o + off]; __syncthreads(); }
  float var = red[0] * (1.f / DM);
  float y = d * rsqrtf(var + 1e-5f) * g[o] + beta[o];
  hf16[(size_t)pos * DM + o] = (_Float16)y;
}

// ---------------------------------------------------------------------------
// Kernel 2: per-chunk attention pool. One wave32 per chunk (4/block).
// t = tanh(H @ ca1w + b) @ ca2w -> softmax(16) -> summ (f16 out)
// ---------------------------------------------------------------------------
__global__ void __launch_bounds__(128)
k_chunk(const _Float16* __restrict__ hf16,
        const float* __restrict__ ca1w, const float* __restrict__ ca1b,
        const float* __restrict__ ca2w, const float* __restrict__ ca2b,
        _Float16* __restrict__ summ16) {
  __shared__ __align__(16) _Float16 shH[4][CHUNK * DM]; // 4KB per wave

  int wave  = threadIdx.x >> 5;
  int lane  = threadIdx.x & 31;
  int chunk = blockIdx.x * 4 + wave;

  // cooperative coalesced load: 16 rows x 128 f16 (contiguous 4KB)
  const uint4* gsrc = (const uint4*)(hf16 + (size_t)chunk * CHUNK * DM);
  uint4* ldst = (uint4*)(&shH[wave][0]);
#pragma unroll
  for (int t = 0; t < 8; ++t) ldst[lane + 32 * t] = gsrc[lane + 32 * t];
  if (chunk + 4 < NCH_TOT)  // global_prefetch_b8 for the next chunk
    __builtin_prefetch(hf16 + (size_t)(chunk + 4) * CHUNK * DM, 0, 1);
  __syncthreads();

  int hi = lane >> 4;   // half-wave group
  int m  = lane & 15;

  Acc acc0, acc1;
#pragma unroll
  for (int r = 0; r < 8; ++r) { acc0.e[r] = 0.f; acc1.e[r] = 0.f; }
#pragma unroll
  for (int ks = 0; ks < 4; ++ks) {
    int kbase = ks * 32;
    AFrag a, b0, b1;
#pragma unroll
    for (int i = 0; i < 16; ++i) {       // A: lane=M, K per ISA 16-bit A layout
      int K = (i < 8 ? i : i + 8) + 8 * hi;
      a.e[i] = shH[wave][m * DM + kbase + K];
    }
#pragma unroll
    for (int i = 0; i < 16; ++i) {       // B: lane=N, K per ISA 16-bit layout
      int Kb = kbase + (i < 8 ? i : i + 8) + 8 * hi;
      b0.e[i] = (_Float16)ca1w[Kb * 32 + m];
      b1.e[i] = (_Float16)ca1w[Kb * 32 + 16 + m];
    }
    acc0.v = __builtin_amdgcn_wmma_f32_16x16x32_f16(false, a.v, false, b0.v, (short)0, acc0.v, false, false);
    acc1.v = __builtin_amdgcn_wmma_f32_16x16x32_f16(false, a.v, false, b1.v, (short)0, acc1.v, false, false);
  }

  // scores sc[16]: D-tile lane holds (M = r + 8*hi, N = m)
  float w0c = ca2w[m], w1c = ca2w[16 + m];
  float b0c = ca1b[m], b1c = ca1b[16 + m];
  float p[8];
#pragma unroll
  for (int r = 0; r < 8; ++r) {
    float t0 = tanhf(acc0.e[r] + b0c);
    float t1 = tanhf(acc1.e[r] + b1c);
    p[r] = t0 * w0c + t1 * w1c;
  }
#pragma unroll
  for (int r = 0; r < 8; ++r) {          // reduce over the 16 N-lanes of this half-wave
    p[r] += __shfl_xor(p[r], 1, 32);
    p[r] += __shfl_xor(p[r], 2, 32);
    p[r] += __shfl_xor(p[r], 4, 32);
    p[r] += __shfl_xor(p[r], 8, 32);
    p[r] += ca2b[0];
  }
  // softmax over the 16 chunk rows (rows 0-7 in hi=0 lanes, 8-15 in hi=1)
  float mx = p[0];
#pragma unroll
  for (int r = 1; r < 8; ++r) mx = fmaxf(mx, p[r]);
  mx = fmaxf(mx, __shfl_xor(mx, 16, 32));
  float e[8]; float sum = 0.f;
#pragma unroll
  for (int r = 0; r < 8; ++r) { e[r] = __expf(p[r] - mx); sum += e[r]; }
  sum += __shfl_xor(sum, 16, 32);
  float inv = 1.f / sum;
  float wrow[16];
#pragma unroll
  for (int r = 0; r < 8; ++r) {
    wrow[r]     = __shfl(e[r], 0, 32)  * inv;
    wrow[8 + r] = __shfl(e[r], 16, 32) * inv;
  }

  // summ[d] = sum_M w[M]*h[M][d] ; lane owns d = 4*lane..4*lane+3; write f16
  float summ[4] = {0.f, 0.f, 0.f, 0.f};
  int d0 = lane * 4;
#pragma unroll
  for (int M = 0; M < 16; ++M) {
    float wM = wrow[M];
#pragma unroll
    for (int k = 0; k < 4; ++k)
      summ[k] += wM * (float)shH[wave][M * DM + d0 + k];
  }
#pragma unroll
  for (int k = 0; k < 4; ++k)
    summ16[(size_t)chunk * DM + d0 + k] = (_Float16)summ[k];
}

// ---------------------------------------------------------------------------
// Kernel 3: batched WMMA projection GEMM.
// ap[:, 0:120]   = summ @ pp_w + pp_b
// ap[:, 120:126] = tanh(summ @ ep_w + ep_b) * pi        (angles)
// One wave32 per 16-row tile; N = 128 (8 tiles), K = 128 (4 steps) -> 32 WMMA.
// ---------------------------------------------------------------------------
__global__ void __launch_bounds__(128)
k_proj(const _Float16* __restrict__ summ16,
       const float* __restrict__ ppw, const float* __restrict__ ppb,
       const float* __restrict__ epw, const float* __restrict__ epb,
       float* __restrict__ apbuf) {
  __shared__ __align__(16) _Float16 shA[4][CHUNK * DM]; // 4KB per wave

  int wave = threadIdx.x >> 5;
  int lane = threadIdx.x & 31;
  int tile = blockIdx.x * 4 + wave;        // 0..1023
  int rowBase = tile * 16;

  const uint4* gsrc = (const uint4*)(summ16 + (size_t)rowBase * DM);
  uint4* ldst = (uint4*)(&shA[wave][0]);
#pragma unroll
  for (int t = 0; t < 8; ++t) ldst[lane + 32 * t] = gsrc[lane + 32 * t];
  __syncthreads();

  int hi = lane >> 4;
  int m  = lane & 15;

  AFrag a[4];
#pragma unroll
  for (int ks = 0; ks < 4; ++ks) {
#pragma unroll
    for (int i = 0; i < 16; ++i) {
      int K = (i < 8 ? i : i + 8) + 8 * hi;
      a[ks].e[i] = shA[wave][m * DM + ks * 32 + K];
    }
  }

#pragma unroll
  for (int n = 0; n < 8; ++n) {            // N-tile: cols n*16 .. n*16+15
    Acc acc;
#pragma unroll
    for (int r = 0; r < 8; ++r) acc.e[r] = 0.f;
#pragma unroll
    for (int ks = 0; ks < 4; ++ks) {
      AFrag b;
#pragma unroll
      for (int i = 0; i < 16; ++i) {
        int dIdx = ks * 32 + (i < 8 ? i : i + 8) + 8 * hi;
        int j = n * 16 + m;
        float wv;
        if (j < 2 * NFP)      wv = ppw[dIdx * (2 * NFP) + j];
        else if (j < 2 * NFP + NQ) wv = epw[dIdx * NQ + (j - 2 * NFP)];
        else                  wv = 0.f;
        b.e[i] = (_Float16)wv;
      }
      acc.v = __builtin_amdgcn_wmma_f32_16x16x32_f16(false, a[ks].v, false, b.v, (short)0, acc.v, false, false);
    }
    // postprocess + store: lane holds (row = r + 8*hi, col j = n*16 + m)
    int j = n * 16 + m;
#pragma unroll
    for (int r = 0; r < 8; ++r) {
      int row = rowBase + r + 8 * hi;
      float v = acc.e[r];
      if (j < 2 * NFP)
        apbuf[(size_t)row * DM + j] = v + ppb[j];
      else if (j < 2 * NFP + NQ)
        apbuf[(size_t)row * DM + j] = tanhf(v + epb[j - 2 * NFP]) * PI_F;
    }
  }
}

// ---------------------------------------------------------------------------
// complex helpers (state vector sim), wave32 lane-exchange via shfl_xor
// ---------------------------------------------------------------------------
__device__ __forceinline__ float2 mkc(float r, float i) { return make_float2(r, i); }
__device__ __forceinline__ float2 cmul(float2 a, float2 b) {
  return make_float2(a.x * b.x - a.y * b.y, a.x * b.y + a.y * b.x);
}
__device__ __forceinline__ float2 cadd(float2 a, float2 b) { return make_float2(a.x + b.x, a.y + b.y); }
__device__ __forceinline__ float2 cconj(float2 a) { return make_float2(a.x, -a.y); }
__device__ __forceinline__ float  cnorm(float2 a) { return a.x * a.x + a.y * a.y; }
__device__ __forceinline__ float2 shx2(float2 a, int m) {
  return make_float2(__shfl_xor(a.x, m, 32), __shfl_xor(a.y, m, 32));
}

// Apply 2x2 (optionally controlled) gate on amplitude bit `tbit`.
// amp index = 2*lane + j ; bit0 = j (in-lane), bit k>=1 = lane bit k-1.
__device__ __forceinline__ void apply1(float2& s0, float2& s1,
                                       float2 u00, float2 u01, float2 u10, float2 u11,
                                       int tbit, int cbit, int lane) {
  bool app0 = true, app1 = true;
  if (cbit == 0) { app0 = false; }
  else if (cbit > 0) { bool c = (lane >> (cbit - 1)) & 1; app0 = c; app1 = c; }
  if (tbit == 0) {
    if (app0 && app1) {
      float2 n0 = cadd(cmul(u00, s0), cmul(u01, s1));
      float2 n1 = cadd(cmul(u10, s0), cmul(u11, s1));
      s0 = n0; s1 = n1;
    }
  } else {
    int mask = 1 << (tbit - 1);
    float2 o0 = shx2(s0, mask);
    float2 o1 = shx2(s1, mask);
    bool hiLane = (lane >> (tbit - 1)) & 1;
    float2 n0, n1;
    if (!hiLane) { n0 = cadd(cmul(u00, s0), cmul(u01, o0)); n1 = cadd(cmul(u00, s1), cmul(u01, o1)); }
    else         { n0 = cadd(cmul(u10, o0), cmul(u11, s0)); n1 = cadd(cmul(u10, o1), cmul(u11, s1)); }
    if (app0) s0 = n0;
    if (app1) s1 = n1;
  }
}

// One ansatz pass (rx,ry,rz per qubit then two CRX rings); qubit w -> bit (5-w)
__device__ void ansatz_pass(float2& s0, float2& s1, const float* p, int& off, int lane) {
  for (int i = 0; i < NQ; ++i) {
    float thx = p[off + 3 * i] * 0.5f;
    float cx = cosf(thx), sx = sinf(thx);
    apply1(s0, s1, mkc(cx, 0), mkc(0, -sx), mkc(0, -sx), mkc(cx, 0), NQ - 1 - i, -1, lane);
    float thy = p[off + 3 * i + 1] * 0.5f;
    float cy = cosf(thy), sy = sinf(thy);
    apply1(s0, s1, mkc(cy, 0), mkc(-sy, 0), mkc(sy, 0), mkc(cy, 0), NQ - 1 - i, -1, lane);
    float thz = p[off + 3 * i + 2] * 0.5f;
    float cz = cosf(thz), sz = sinf(thz);
    apply1(s0, s1, mkc(cz, -sz), mkc(0, 0), mkc(0, 0), mkc(cz, sz), NQ - 1 - i, -1, lane);
  }
  off += 3 * NQ;
  for (int i = 0; i < NQ; ++i) {
    float th = p[off] * 0.5f; off++;
    float c = cosf(th), s = sinf(th);
    apply1(s0, s1, mkc(c, 0), mkc(0, -s), mkc(0, -s), mkc(c, 0),
           NQ - 1 - ((i + 1) % NQ), NQ - 1 - i, lane);
  }
  for (int i = NQ - 1; i >= 0; --i) {
    float th = p[off] * 0.5f; off++;
    float c = cosf(th), s = sinf(th);
    apply1(s0, s1, mkc(c, 0), mkc(0, -s), mkc(0, -s), mkc(c, 0),
           NQ - 1 - ((i + NQ - 1) % NQ), NQ - 1 - i, lane);
  }
}

// ---------------------------------------------------------------------------
// Kernel 4: 6-qubit circuit, one wave32 per chunk (64 amps, 2/lane)
// ---------------------------------------------------------------------------
__global__ void __launch_bounds__(128)
k_quantum(const float* __restrict__ apbuf, float* __restrict__ qout) {
  __shared__ __align__(16) float shP[4][DM];  // fp 0..59 | mp 60..119 | ang 120..125
  int wave  = threadIdx.x >> 5;
  int lane  = threadIdx.x & 31;
  int chunk = blockIdx.x * 4 + wave;

  const float4* src = (const float4*)(apbuf + (size_t)chunk * DM);
  ((float4*)&shP[wave][0])[lane] = src[lane];
  __syncthreads();

  float2 s0 = make_float2(lane == 0 ? 1.f : 0.f, 0.f);
  float2 s1 = make_float2(0.f, 0.f);
  const float* fp  = shP[wave];
  const float* mp  = shP[wave] + NFP;
  const float* ang = shP[wave] + 2 * NFP;
  for (int i = 0; i < NQ; ++i) {         // initial RY(ang[i]) on qubit i
    float th = ang[i] * 0.5f;
    float c = cosf(th), s = sinf(th);
    apply1(s0, s1, mkc(c, 0), mkc(-s, 0), mkc(s, 0), mkc(c, 0), NQ - 1 - i, -1, lane);
  }
  int off = 0;
  for (int r = 0; r < 2; ++r) ansatz_pass(s0, s1, fp, off, lane);
  off = 0;
  for (int r = 0; r < 2; ++r) ansatz_pass(s0, s1, mp, off, lane);

  // expectations: qubit i -> amp bit b = 5-i
  for (int i = 0; i < NQ; ++i) {
    int b = NQ - 1 - i;
    float2 ab = make_float2(0.f, 0.f); float zz = 0.f;
    if (b == 0) {
      ab = cmul(cconj(s0), s1);
      zz = cnorm(s0) - cnorm(s1);
    } else {
      int mask = 1 << (b - 1);
      float2 p0 = shx2(s0, mask), p1 = shx2(s1, mask);
      bool lo = (((lane >> (b - 1)) & 1) == 0);
      if (lo) {
        ab = cadd(cmul(cconj(s0), p0), cmul(cconj(s1), p1));
        zz = cnorm(s0) + cnorm(s1) - cnorm(p0) - cnorm(p1);
      }
    }
#pragma unroll
    for (int msk = 1; msk < 32; msk <<= 1) {
      ab.x += __shfl_xor(ab.x, msk, 32);
      ab.y += __shfl_xor(ab.y, msk, 32);
      zz   += __shfl_xor(zz,   msk, 32);
    }
    if (lane == 0) {
      qout[(size_t)chunk * QOUT + i]          = 2.f * ab.x;
      qout[(size_t)chunk * QOUT + NQ + i]     = 2.f * ab.y;
      qout[(size_t)chunk * QOUT + 2 * NQ + i] = zz;
    }
  }
}

// ---------------------------------------------------------------------------
// Kernel 5: cf = silu( LN( q @ op_w + op_b ) ) ; block per chunk row
// ---------------------------------------------------------------------------
__global__ void k_qproj(const float* __restrict__ q, const float* __restrict__ opw,
                        const float* __restrict__ opb, const float* __restrict__ g,
                        const float* __restrict__ beta, float* __restrict__ cf) {
  int row = blockIdx.x; int j = threadIdx.x;
  float v = opb[j];
#pragma unroll
  for (int k = 0; k < QOUT; ++k) v += q[(size_t)row * QOUT + k] * opw[k * DM + j];
  __shared__ float red[DM];
  red[j] = v; __syncthreads();
  for (int off = 64; off > 0; off >>= 1) { if (j < off) red[j] += red[j + off]; __syncthreads(); }
  float mean = red[0] * (1.f / DM);
  __syncthreads();
  float d = v - mean;
  red[j] = d * d; __syncthreads();
  for (int off = 64; off > 0; off >>= 1) { if (j < off) red[j] += red[j + off]; __syncthreads(); }
  float var = red[0] * (1.f / DM);
  float y = d * rsqrtf(var + 1e-5f) * g[j] + beta[j];
  y = y / (1.f + __expf(-y));  // silu
  cf[(size_t)row * DM + j] = y;
}

// ---------------------------------------------------------------------------
// Kernel 6: sequence attention + classifier. One block (128 thr) per batch.
// ---------------------------------------------------------------------------
__global__ void k_seq(const float* __restrict__ cf,
                      const float* __restrict__ sa1w, const float* __restrict__ sa1b,
                      const float* __restrict__ sa2w, const float* __restrict__ sa2b,
                      const float* __restrict__ cl1w, const float* __restrict__ cl1b,
                      const float* __restrict__ cl2w, const float* __restrict__ cl2b,
                      float* __restrict__ out) {
  int b = blockIdx.x; int t = threadIdx.x;   // t = chunk index for scoring
  const float* cfb = cf + (size_t)b * NCHUNK * DM;
  float sk[32];
#pragma unroll
  for (int k = 0; k < 32; ++k) sk[k] = sa1b[k];
  for (int d = 0; d < DM; ++d) {
    float v = cfb[(size_t)t * DM + d];
#pragma unroll
    for (int k = 0; k < 32; ++k) sk[k] += v * sa1w[d * 32 + k];
  }
  float sc = sa2b[0];
#pragma unroll
  for (int k = 0; k < 32; ++k) sc += tanhf(sk[k]) * sa2w[k];

  __shared__ float red[128];
  __shared__ float sw[128];
  __shared__ float rep[128];
  __shared__ float uu[64];
  red[t] = sc; __syncthreads();
  for (int off = 64; off > 0; off >>= 1) { if (t < off) red[t] = fmaxf(red[t], red[t + off]); __syncthreads(); }
  float mx = red[0]; __syncthreads();
  float e = __expf(sc - mx);
  red[t] = e; __syncthreads();
  for (int off = 64; off > 0; off >>= 1) { if (t < off) red[t] += red[t + off]; __syncthreads(); }
  float tot = red[0];
  sw[t] = e / tot; __syncthreads();

  float r = 0.f;                      // rep[d], d = t
  for (int c = 0; c < NCHUNK; ++c) r += sw[c] * cfb[(size_t)c * DM + t];
  rep[t] = r; __syncthreads();

  if (t < 64) {
    float u = cl1b[t];
    for (int d = 0; d < DM; ++d) u += rep[d] * cl1w[d * 64 + t];
    u = u / (1.f + __expf(-u));       // silu
    uu[t] = u;
  }
  __syncthreads();
  if (t < 2) {
    float o = cl2b[t];
#pragma unroll
    for (int j = 0; j < 64; ++j) o += uu[j] * cl2w[j * 2 + t];
    out[b * 2 + t] = o;
  }
}

// ---------------------------------------------------------------------------
extern "C" void kernel_launch(void* const* d_in, const int* in_sizes, int n_in,
                              void* d_out, int out_size, void* d_ws, size_t ws_size,
                              hipStream_t stream) {
  (void)in_sizes; (void)n_in; (void)out_size; (void)ws_size;
  const float* x     = (const float*)d_in[0];
  const float* convw = (const float*)d_in[1];
  const float* convb = (const float*)d_in[2];
  const float* ln1g  = (const float*)d_in[3];
  const float* ln1b  = (const float*)d_in[4];
  const float* ca1w  = (const float*)d_in[5];
  const float* ca1b  = (const float*)d_in[6];
  const float* ca2w  = (const float*)d_in[7];
  const float* ca2b  = (const float*)d_in[8];
  const float* ppw   = (const float*)d_in[9];
  const float* ppb   = (const float*)d_in[10];
  const float* epw   = (const float*)d_in[11];
  const float* epb   = (const float*)d_in[12];
  const float* opw   = (const float*)d_in[13];
  const float* opb   = (const float*)d_in[14];
  const float* ln2g  = (const float*)d_in[15];
  const float* ln2b  = (const float*)d_in[16];
  const float* sa1w  = (const float*)d_in[17];
  const float* sa1b  = (const float*)d_in[18];
  const float* sa2w  = (const float*)d_in[19];
  const float* sa2b  = (const float*)d_in[20];
  const float* cl1w  = (const float*)d_in[21];
  const float* cl1b  = (const float*)d_in[22];
  const float* cl2w  = (const float*)d_in[23];
  const float* cl2b  = (const float*)d_in[24];

  char* ws = (char*)d_ws;
  size_t off_h    = 0;                                                   // 64 MB
  size_t off_summ = off_h    + (size_t)NROWS   * DM   * sizeof(_Float16); // +4 MB
  size_t off_ap   = off_summ + (size_t)NCH_TOT * DM   * sizeof(_Float16); // +8 MB
  size_t off_q    = off_ap   + (size_t)NCH_TOT * DM   * sizeof(float);    // +1.1 MB
  size_t off_cf   = off_q    + (size_t)NCH_TOT * QOUT * sizeof(float);    // +8 MB
  _Float16* hf16   = (_Float16*)(ws + off_h);
  _Float16* summ16 = (_Float16*)(ws + off_summ);
  float*    apbuf  = (float*)(ws + off_ap);
  float*    qbuf   = (float*)(ws + off_q);
  float*    cfbuf  = (float*)(ws + off_cf);

  k_conv_ln<<<NROWS, DM, 0, stream>>>(x, convw, convb, ln1g, ln1b, hf16);
  k_chunk  <<<NCH_TOT / 4, 128, 0, stream>>>(hf16, ca1w, ca1b, ca2w, ca2b, summ16);
  k_proj   <<<NCH_TOT / 16 / 4, 128, 0, stream>>>(summ16, ppw, ppb, epw, epb, apbuf);
  k_quantum<<<NCH_TOT / 4, 128, 0, stream>>>(apbuf, qbuf);
  k_qproj  <<<NCH_TOT, DM, 0, stream>>>(qbuf, opw, opb, ln2g, ln2b, cfbuf);
  k_seq    <<<BATCH, 128, 0, stream>>>(cfbuf, sa1w, sa1b, sa2w, sa2b,
                                       cl1w, cl1b, cl2w, cl2b, (float*)d_out);
}
